// BaichuanAttention_53790170415192
// MI455X (gfx1250) — matrix operand order repaired and checked by reading, not compiled, not run on hardware
//
#include <hip/hip_runtime.h>
#include <hip/hip_bf16.h>

typedef __attribute__((ext_vector_type(16))) __bf16        v16bf;
typedef __attribute__((ext_vector_type(8)))  float         v8f;
typedef __attribute__((ext_vector_type(4)))  unsigned int  u32x4;

#define HIDN  4096
#define NHEAD 32
#define HDIM  128
#define BATCH 2
#define SEQ   1024
#define MTOT  (BATCH*SEQ)

__device__ __forceinline__ v16bf frag2(u32x4 a, u32x4 b) {
  union { u32x4 u[2]; v16bf v; } x;
  x.u[0] = a; x.u[1] = b;
  return x.v;
}

__device__ __forceinline__ v8f wmma_bf16(v16bf a, v16bf b, v8f c) {
  // D = A(16x32 bf16) x B(32x16 bf16) + C(16x16 f32)
  return __builtin_amdgcn_wmma_f32_16x16x32_bf16(
      false, a, false, b, (short)0, c, false, false);
}

// ---------------------------------------------------------------- convert
__global__ void cvt_f32_bf16(const float* __restrict__ in,
                             __bf16* __restrict__ out, int n) {
  int i = blockIdx.x * blockDim.x + threadIdx.x;
  int stride = gridDim.x * blockDim.x;
  for (; i < n; i += stride) out[i] = (__bf16)in[i];
}

// ---------------------------------------------------------------- GEMM
// C[M,N] = A[M,K](bf16,row-major) * W[N,K](bf16,row-major)^T
// Block: 256 thr (8 waves), tile 128x128, Kc=32, double-buffered LDS.
// Wave tile 64x32 (4x2 wmma). K is compile-time for cheap addressing.
#define LDST 40   // LDS row stride in bf16 elems (80B: 16B-aligned, padded)

template <int K>
__global__ __launch_bounds__(256)
void gemm_bf16_tn(const __bf16* __restrict__ A, const __bf16* __restrict__ W,
                  float* __restrict__ Cf, __bf16* __restrict__ Cb, int N) {
  __shared__ __bf16 smA[2][128 * LDST];
  __shared__ __bf16 smW[2][128 * LDST];
  const int tid  = threadIdx.x;
  const int lane = tid & 31;
  const int wid  = tid >> 5;
  const int wm   = wid >> 2;          // 0..1
  const int wn   = wid & 3;           // 0..3
  const int mbase = blockIdx.y * 128;
  const int nbase = blockIdx.x * 128;
  const int hi   = (lane < 16) ? 0 : 1;
  const int l15  = lane & 15;

  // per-thread staging: 2 x 16B chunks per matrix; chunk2 = chunk1 + 64 rows
  const __bf16* ga = A + (size_t)(mbase + (tid >> 2)) * K + (tid & 3) * 8;
  const __bf16* gw = W + (size_t)(nbase + (tid >> 2)) * K + (tid & 3) * 8;
  const int soff = (tid >> 2) * LDST + (tid & 3) * 8;

  // per-wave fragment read offsets (constant across K loop)
  int aoff[4], woff[2];
#pragma unroll
  for (int mi = 0; mi < 4; mi++)
    aoff[mi] = (wm * 64 + mi * 16 + l15) * LDST + hi * 8;
#pragma unroll
  for (int ni = 0; ni < 2; ni++)
    woff[ni] = (wn * 32 + ni * 16 + l15) * LDST + hi * 8;

  v8f acc[4][2];
#pragma unroll
  for (int i = 0; i < 4; i++)
#pragma unroll
    for (int j = 0; j < 2; j++)
#pragma unroll
      for (int r = 0; r < 8; r++) acc[i][j][r] = 0.0f;

  auto compute = [&](int buf) {
    const __bf16* bA = smA[buf];
    const __bf16* bW = smW[buf];
    v16bf af[4], bf_[2];
#pragma unroll
    for (int mi = 0; mi < 4; mi++) {
      const __bf16* p = bA + aoff[mi];
      af[mi] = frag2(*(const u32x4*)p, *(const u32x4*)(p + 16));
    }
#pragma unroll
    for (int ni = 0; ni < 2; ni++) {
      const __bf16* p = bW + woff[ni];
      bf_[ni] = frag2(*(const u32x4*)p, *(const u32x4*)(p + 16));
    }
#pragma unroll
    for (int mi = 0; mi < 4; mi++)
#pragma unroll
      for (int ni = 0; ni < 2; ni++)
        acc[mi][ni] = wmma_bf16(af[mi], bf_[ni], acc[mi][ni]);
  };

  // prologue: stage tile 0 into buf 0
  {
    u32x4 a0 = *(const u32x4*)(ga);
    u32x4 a1 = *(const u32x4*)(ga + 64 * K);
    u32x4 w0 = *(const u32x4*)(gw);
    u32x4 w1 = *(const u32x4*)(gw + 64 * K);
    *(u32x4*)(&smA[0][soff])             = a0;
    *(u32x4*)(&smA[0][soff + 64 * LDST]) = a1;
    *(u32x4*)(&smW[0][soff])             = w0;
    *(u32x4*)(&smW[0][soff + 64 * LDST]) = w1;
  }
  __syncthreads();

  int buf = 0;
#pragma unroll 1
  for (int kb = 0; kb < K - 32; kb += 32) {
    // issue next tile's global loads first (latency hidden behind compute)
    u32x4 a0 = *(const u32x4*)(ga + kb + 32);
    u32x4 a1 = *(const u32x4*)(ga + kb + 32 + 64 * K);
    u32x4 w0 = *(const u32x4*)(gw + kb + 32);
    u32x4 w1 = *(const u32x4*)(gw + kb + 32 + 64 * K);

    compute(buf);

    int nb = buf ^ 1;
    *(u32x4*)(&smA[nb][soff])             = a0;
    *(u32x4*)(&smA[nb][soff + 64 * LDST]) = a1;
    *(u32x4*)(&smW[nb][soff])             = w0;
    *(u32x4*)(&smW[nb][soff + 64 * LDST]) = w1;
    __syncthreads();
    buf = nb;
  }
  compute(buf);   // last tile, no staging

  // epilogue: C layout comp r -> row r (lanes 0-15) / r+8 (lanes 16-31), col = lane&15
#pragma unroll
  for (int mi = 0; mi < 4; mi++)
#pragma unroll
    for (int ni = 0; ni < 2; ni++) {
      int col = nbase + wn * 32 + ni * 16 + l15;
#pragma unroll
      for (int r = 0; r < 8; r++) {
        int row = mbase + wm * 64 + mi * 16 + r + hi * 8;
        size_t off = (size_t)row * N + col;
        if (Cb) Cb[off] = (__bf16)acc[mi][ni][r];
        else    Cf[off] = acc[mi][ni][r];
      }
    }
}

// ---------------------------------------------------------------- RoPE + repack
// qkv bf16 [M, 3*HIDN] -> q,k bf16 [B,NH,S,HD] (rope'd; q pre-scaled by HD^-0.5)
//                         v  bf16 [B,NH,HD,S] (transposed for PV B-operand)
__global__ void rope_split(const __bf16* __restrict__ qkv,
                           const int* __restrict__ pos,
                           __bf16* __restrict__ qo, __bf16* __restrict__ ko,
                           __bf16* __restrict__ vt) {
  int idx = blockIdx.x * blockDim.x + threadIdx.x;   // B*S*NH*64 threads
  int d = idx & 63;
  int h = (idx >> 6) & (NHEAD - 1);
  int s = (idx >> 11) & (SEQ - 1);
  int b = idx >> 21;
  if (b >= BATCH) return;
  int m = b * SEQ + s;
  const __bf16* row = qkv + (size_t)m * (3 * HIDN);
  float q1 = (float)row[h * HDIM + d];
  float q2 = (float)row[h * HDIM + d + 64];
  float k1 = (float)row[HIDN + h * HDIM + d];
  float k2 = (float)row[HIDN + h * HDIM + d + 64];
  float v1 = (float)row[2 * HIDN + h * HDIM + d];
  float v2 = (float)row[2 * HIDN + h * HDIM + d + 64];

  float p = (float)pos[m];
  float freq = p * __powf(10000.0f, -(float)d * (1.0f / 64.0f));
  float c = cosf(freq), sn = sinf(freq);
  const float scale = 0.08838834764831845f;   // 1/sqrt(128)

  size_t qkoff = ((size_t)(b * NHEAD + h) * SEQ + s) * HDIM + d;
  qo[qkoff]      = (__bf16)((q1 * c - q2 * sn) * scale);
  qo[qkoff + 64] = (__bf16)((q2 * c + q1 * sn) * scale);
  ko[qkoff]      = (__bf16)(k1 * c - k2 * sn);
  ko[qkoff + 64] = (__bf16)(k2 * c + k1 * sn);

  size_t vtoff = ((size_t)(b * NHEAD + h) * HDIM + d) * SEQ + s;
  vt[vtoff]            = (__bf16)v1;
  vt[vtoff + 64 * SEQ] = (__bf16)v2;
}

// ---------------------------------------------------------------- attention
// One block = 8 waves, each wave owns a 16-row q tile; causal flash attention.
__global__ __launch_bounds__(256)
void attn_kernel(const __bf16* __restrict__ q, const __bf16* __restrict__ k,
                 const __bf16* __restrict__ vt, __bf16* __restrict__ attn) {
  __shared__ __bf16 pbuf[8][16 * 32];   // per-wave P re-layout scratch
  const int lane = threadIdx.x & 31;
  const int wid  = threadIdx.x >> 5;
  const int hi   = (lane < 16) ? 0 : 1;
  const int l15  = lane & 15;
  const int nqb  = SEQ / 128;
  int bh   = blockIdx.x / nqb;
  int qblk = blockIdx.x % nqb;
  int b = bh / NHEAD, h = bh % NHEAD;
  int q0 = qblk * 128 + wid * 16;

  const __bf16* qb = q  + (size_t)bh * SEQ * HDIM;
  const __bf16* kb = k  + (size_t)bh * SEQ * HDIM;
  const __bf16* vb = vt + (size_t)bh * HDIM * SEQ;

  // q A-fragments: 16 rows x 128 dims (4 k-chunks of 32)
  v16bf qf[4];
  {
    const __bf16* p = qb + (size_t)(q0 + l15) * HDIM;
#pragma unroll
    for (int c2 = 0; c2 < 4; c2++) {
      const __bf16* pc = p + c2 * 32 + hi * 8;
      qf[c2] = frag2(*(const u32x4*)pc, *(const u32x4*)(pc + 16));
    }
  }

  v8f o[8], mrun, lrun;
#pragma unroll
  for (int i = 0; i < 8; i++)
#pragma unroll
    for (int r = 0; r < 8; r++) o[i][r] = 0.0f;
#pragma unroll
  for (int r = 0; r < 8; r++) { mrun[r] = -1e30f; lrun[r] = 0.0f; }

  int jend = q0 + 16;            // causal: keys < q0+16
  for (int jb = 0; jb < jend; jb += 32) {
    // scores 16x32 = q(16x128) . k^T  (two 16-col tiles, 4 K-chunks)
    v8f s[2];
#pragma unroll
    for (int t = 0; t < 2; t++)
#pragma unroll
      for (int r = 0; r < 8; r++) s[t][r] = 0.0f;
#pragma unroll
    for (int t = 0; t < 2; t++) {
      const __bf16* kr = kb + (size_t)(jb + t * 16 + l15) * HDIM;
#pragma unroll
      for (int c2 = 0; c2 < 4; c2++) {
        const __bf16* pc = kr + c2 * 32 + hi * 8;
        v16bf bfrag = frag2(*(const u32x4*)pc, *(const u32x4*)(pc + 16));
        s[t] = wmma_bf16(qf[c2], bfrag, s[t]);
      }
    }
    // causal mask (scale already folded into q)
#pragma unroll
    for (int t = 0; t < 2; t++) {
      int j = jb + t * 16 + l15;
#pragma unroll
      for (int r = 0; r < 8; r++) {
        int qpos = q0 + r + hi * 8;
        if (j > qpos) s[t][r] = -1e30f;
      }
    }
    // online softmax: per-row state replicated across each 16-lane group
    v8f mnew, alpha, p0, p1;
#pragma unroll
    for (int r = 0; r < 8; r++) {
      float mx = fmaxf(s[0][r], s[1][r]);
#pragma unroll
      for (int off = 1; off < 16; off <<= 1) mx = fmaxf(mx, __shfl_xor(mx, off, 32));
      mnew[r]  = fmaxf(mrun[r], mx);
      alpha[r] = __expf(mrun[r] - mnew[r]);
      mrun[r]  = mnew[r];
    }
#pragma unroll
    for (int r = 0; r < 8; r++) {
      p0[r] = __expf(s[0][r] - mnew[r]);
      p1[r] = __expf(s[1][r] - mnew[r]);
      float rs = p0[r] + p1[r];
#pragma unroll
      for (int off = 1; off < 16; off <<= 1) rs += __shfl_xor(rs, off, 32);
      lrun[r] = lrun[r] * alpha[r] + rs;
    }
#pragma unroll
    for (int ni = 0; ni < 8; ni++)
#pragma unroll
      for (int r = 0; r < 8; r++) o[ni][r] *= alpha[r];

    // re-layout P (C-layout -> A-operand) via per-wave LDS scratch
#pragma unroll
    for (int r = 0; r < 8; r++) {
      int row = r + hi * 8;
      pbuf[wid][row * 32 + l15]      = (__bf16)p0[r];
      pbuf[wid][row * 32 + 16 + l15] = (__bf16)p1[r];
    }
    const __bf16* pp = &pbuf[wid][l15 * 32 + hi * 8];
    v16bf pf = frag2(*(const u32x4*)pp, *(const u32x4*)(pp + 16));

    // o(16x128) += P(16x32) . V(32x128)   (V transposed: per-lane contiguous)
#pragma unroll
    for (int ni = 0; ni < 8; ni++) {
      const __bf16* vr = vb + (size_t)(ni * 16 + l15) * SEQ + jb + hi * 8;
      v16bf vfrag = frag2(*(const u32x4*)vr, *(const u32x4*)(vr + 16));
      o[ni] = wmma_bf16(pf, vfrag, o[ni]);
    }
  }

  // finalize + store bf16 [M, HIDN] for the o_proj GEMM
  v8f inv;
#pragma unroll
  for (int r = 0; r < 8; r++) inv[r] = 1.0f / lrun[r];
#pragma unroll
  for (int ni = 0; ni < 8; ni++) {
    int col = h * HDIM + ni * 16 + l15;
#pragma unroll
    for (int r = 0; r < 8; r++) {
      int row = b * SEQ + q0 + r + hi * 8;
      attn[(size_t)row * HIDN + col] = (__bf16)(o[ni][r] * inv[r]);
    }
  }
}

// ---------------------------------------------------------------- launch
extern "C" void kernel_launch(void* const* d_in, const int* in_sizes, int n_in,
                              void* d_out, int out_size, void* d_ws, size_t ws_size,
                              hipStream_t stream) {
  const int*   positions = (const int*)d_in[0];
  const float* hidden    = (const float*)d_in[1];
  const float* Wpack     = (const float*)d_in[2];
  const float* Oproj     = (const float*)d_in[3];
  float*       out       = (float*)d_out;

  char* p = (char*)d_ws;
  __bf16* h_bf   = (__bf16*)p; p += (size_t)MTOT * HIDN * 2;      // dead after QKV GEMM
  __bf16* wp_bf  = (__bf16*)p; p += (size_t)3 * HIDN * HIDN * 2;
  __bf16* op_bf  = (__bf16*)p; p += (size_t)HIDN * HIDN * 2;
  __bf16* qkv_bf = (__bf16*)p; p += (size_t)MTOT * 3 * HIDN * 2;
  __bf16* q_bf   = (__bf16*)p; p += (size_t)MTOT * HIDN * 2;
  __bf16* k_bf   = (__bf16*)p; p += (size_t)MTOT * HIDN * 2;
  __bf16* vt_bf  = (__bf16*)p; p += (size_t)MTOT * HIDN * 2;
  __bf16* attn_bf = h_bf;                                         // alias (safe ordering)

  cvt_f32_bf16<<<2048, 256, 0, stream>>>(hidden, h_bf, MTOT * HIDN);
  cvt_f32_bf16<<<4096, 256, 0, stream>>>(Wpack, wp_bf, 3 * HIDN * HIDN);
  cvt_f32_bf16<<<2048, 256, 0, stream>>>(Oproj, op_bf, HIDN * HIDN);

  dim3 g1(3 * HIDN / 128, MTOT / 128);
  gemm_bf16_tn<HIDN><<<g1, 256, 0, stream>>>(h_bf, wp_bf, nullptr, qkv_bf,
                                             3 * HIDN);

  rope_split<<<(BATCH * SEQ * NHEAD * 64) / 256, 256, 0, stream>>>(
      qkv_bf, positions, q_bf, k_bf, vt_bf);

  attn_kernel<<<BATCH * NHEAD * (SEQ / 128), 256, 0, stream>>>(
      q_bf, k_bf, vt_bf, attn_bf);

  dim3 g2(HIDN / 128, MTOT / 128);
  gemm_bf16_tn<HIDN><<<g2, 256, 0, stream>>>(attn_bf, op_bf, out, nullptr,
                                             HIDN);
}
